// DAMIC_22033182228939
// MI455X (gfx1250) — matrix operand order repaired and checked by reading, not compiled, not run on hardware
//
#include <hip/hip_runtime.h>
#include <hip/hip_bf16.h>

typedef __attribute__((ext_vector_type(16))) _Float16 v16h;
typedef __attribute__((ext_vector_type(8)))  _Float16 v8h;
typedef __attribute__((ext_vector_type(4)))  _Float16 v4h;
typedef __attribute__((ext_vector_type(8)))  float    v8f;

#define WMMA(a,b,c) __builtin_amdgcn_wmma_f32_16x16x32_f16(false,(a),false,(b),(short)0,(c),false,false)

__device__ __forceinline__ float sigf(float x){ return 1.0f/(1.0f + expf(-x)); }

// ---------------------------------------------------------------------------
// Fragment loaders (CDNA5 wave32 WMMA layouts, CDNA5 ISA §7.12.2)
// A (16x32 f16): lane holds row m=lane%16; element e -> k = e + 8*(e/8) + 8*(lane/16)
//   => two contiguous 8-half runs per lane: [kb, kb+8) and [kb+16, kb+24), kb = k0 + 8*(lane/16)
// B (32x16 f16): lane holds col n=lane%16; element e -> k = e + 8*(e/8) + 8*(lane/16) (same order as A)
// C/D (16x16 f32): element r -> row m = r + 8*(lane/16), col n = lane%16
// Loaders take a precomputed per-lane base = &A[m][kb]; callers bump the
// pointer by a constant per k-step so the k-loop is pure pointer increments.
// ---------------------------------------------------------------------------

// 16-byte-aligned base (row stride % 8 == 0): two b128 loads
__device__ __forceinline__ v16h frag_a16(const _Float16* base){
  v8h lo = *(const v8h*)(base);
  v8h hi = *(const v8h*)(base + 16);
  return __builtin_shufflevector(lo, hi, 0,1,2,3,4,5,6,7,8,9,10,11,12,13,14,15);
}

// 8-byte-aligned base (row stride % 4 == 0, e.g. 300): four b64 loads
__device__ __forceinline__ v16h frag_a8(const _Float16* base){
  v4h a0 = *(const v4h*)(base);
  v4h a1 = *(const v4h*)(base + 4);
  v4h a2 = *(const v4h*)(base + 16);
  v4h a3 = *(const v4h*)(base + 20);
  v8h lo = __builtin_shufflevector(a0, a1, 0,1,2,3,4,5,6,7);
  v8h hi = __builtin_shufflevector(a2, a3, 0,1,2,3,4,5,6,7);
  return __builtin_shufflevector(lo, hi, 0,1,2,3,4,5,6,7,8,9,10,11,12,13,14,15);
}

// B pre-packed per (ktile,ntile) tile, lane-major: Bp[((kt*ntiles+nt)*32 + lane)*16 + e]
__device__ __forceinline__ v16h frag_b(const _Float16* base){
  return *(const v16h*)(base);
}
__device__ __forceinline__ const _Float16* b_base(const _Float16* Bp, int lane, int nt, int ntiles){
  (void)ntiles;
  return Bp + (size_t)((nt << 5) + lane) * 16;
}

// ---------------------------------------------------------------------------
// Weight prep: f32 row-major [N][K] -> f16 fragment-packed B tiles (zero pad)
// ---------------------------------------------------------------------------
__global__ __launch_bounds__(256) void prep_b_packed(const float* __restrict__ src,
                                                     _Float16* __restrict__ dst,
                                                     int N, int K, int ktiles, int ntiles){
  int idx = blockIdx.x * blockDim.x + threadIdx.x;
  int total = ktiles * ntiles * 512;
  if (idx >= total) return;
  int e    = idx & 15;
  int lane = (idx >> 4) & 31;
  int tile = idx >> 9;
  int kt = tile / ntiles, nt = tile - kt*ntiles;
  int k = kt*32 + (e & 7) + ((lane >> 4) << 3) + ((e >> 3) << 4);
  int n = nt*16 + (lane & 15);
  float v = (k < K && n < N) ? src[n*K + k] : 0.0f;
  dst[idx] = (_Float16)v;
}

// ---------------------------------------------------------------------------
// Conv + ReLU + max-over-time. One block per sentence (B*T = 1600 blocks).
// im2col identity: A[p][k] = x_flat[p*300 + k] (window is a contiguous slice
// of the flattened sentence). k-tile tails past K are neutralized by the
// zero-padded packed B; rows p >= P are never consumed by the maxpool.
// Dual accumulator chains (even/odd k-tiles) keep consecutive v_wmma ops
// independent so hazard slots are filled with the other chain's work.
// ---------------------------------------------------------------------------
__global__ __launch_bounds__(256) void conv_embed_kernel(
    const int* __restrict__ dialogue, const float* __restrict__ emb,
    const _Float16* __restrict__ B3, const _Float16* __restrict__ B4, const _Float16* __restrict__ B5,
    const float* __restrict__ cb3, const float* __restrict__ cb4, const float* __restrict__ cb5,
    _Float16* __restrict__ r_in_h)
{
  __shared__ _Float16 x_lds[15616];
  __shared__ int feat[300];
  __shared__ int toks[50];
  const int s = blockIdx.x;
  const int tid = threadIdx.x;

  if (tid < 50) toks[tid] = dialogue[s*50 + tid];
  if (tid < 300) feat[tid] = 0;           // == 0.0f, ReLU floor
  __syncthreads();

  for (int i = tid; i < 15616; i += 256) {
    _Float16 v = (_Float16)0.0f;
    if (i < 15000) {
      int t = i / 300;
      int d = i - t*300;
      v = (_Float16)emb[(size_t)toks[t]*300 + d];
    }
    x_lds[i] = v;
  }
  __syncthreads();

  const int wave = tid >> 5, lane = tid & 31;
  for (int job = wave; job < 63; job += 8) {
    int fsIdx = job / 21;
    int rem   = job - fsIdx*21;
    int mtile = rem / 7;
    int ntile = rem - mtile*7;
    int fs = 3 + fsIdx;
    int P  = 50 - fs + 1;
    int ktiles = (fs*300 + 31) >> 5;
    const _Float16* Bw = (fsIdx == 0) ? B3 : (fsIdx == 1 ? B4 : B5);
    const float*    cb = (fsIdx == 0) ? cb3 : (fsIdx == 1 ? cb4 : cb5);

    // per-lane fragment base pointers; bump by constant per k-tile
    const _Float16* ap = x_lds + (mtile*16 + (lane & 15))*300 + ((lane >> 4) << 3);
    const _Float16* bp = b_base(Bw, lane, ntile, 7);
    const size_t bstep = (size_t)7 * 512;          // one k-tile of packed B

    v8f accA = {}, accB = {};
    int kt = 0;
    for (; kt + 1 < ktiles; kt += 2) {
      v16h a0 = frag_a8(ap);
      v16h b0 = frag_b(bp);
      v16h a1 = frag_a8(ap + 32);
      v16h b1 = frag_b(bp + bstep);
      accA = WMMA(a0, b0, accA);
      accB = WMMA(a1, b1, accB);
      ap += 64;
      bp += 2*bstep;
    }
    if (kt < ktiles) accA = WMMA(frag_a8(ap), frag_b(bp), accA);

    int n = ntile*16 + (lane & 15);
    if (n < 100) {
      float bias = cb[n];
      float mx = 0.0f;
      if (mtile < 2) {
#pragma unroll
        for (int r = 0; r < 8; ++r) mx = fmaxf(mx, accA[r] + accB[r] + bias);
      } else {
#pragma unroll
        for (int r = 0; r < 8; ++r) {
          int p = 32 + r + ((lane >> 4) << 3);
          if (p < P) mx = fmaxf(mx, accA[r] + accB[r] + bias);
        }
      }
      atomicMax(&feat[fsIdx*100 + n], __float_as_int(mx));
    }
  }
  __syncthreads();

  for (int i = tid; i < 320; i += 256) {
    float v = (i < 300) ? __int_as_float(feat[i]) : 0.0f;
    r_in_h[(size_t)s*320 + i] = (_Float16)v;
  }
}

// ---------------------------------------------------------------------------
// LSTM: grid.x = 2 directions, one 1024-thread block (32 waves) each.
// Per step: G = Xt@Wih^T + H@Whh^T + b as WMMA tiles. Each wave owns two
// (mtile,utile) pairs and computes all 4 gate tiles for each, so the lane
// holding C(m,u) has i/f/g/o locally. c-state lives in registers; h in LDS f16.
// dir 0 ("rev", weights wih/whh): iterates t = 24..0; dir 1 ("fwd"): t = 0..24.
// The 4 gate accumulators give distance-4 WMMA chains (no hazard stalls).
// ---------------------------------------------------------------------------
__global__ __launch_bounds__(1024) void lstm_kernel(
    const _Float16* __restrict__ r_in_h,
    const _Float16* __restrict__ Bih0, const _Float16* __restrict__ Bhh0,
    const float* __restrict__ bih0, const float* __restrict__ bhh0,
    const _Float16* __restrict__ Bih1, const _Float16* __restrict__ Bhh1,
    const float* __restrict__ bih1, const float* __restrict__ bhh1,
    _Float16* __restrict__ hs)
{
  const int dir = blockIdx.x;
  const _Float16* Bi = dir ? Bih1 : Bih0;
  const _Float16* Bh = dir ? Bhh1 : Bhh0;
  const float* bi = dir ? bih1 : bih0;
  const float* bh = dir ? bhh1 : bhh0;

  const int tid = threadIdx.x;
  const int wave = tid >> 5, lane = tid & 31;

  __shared__ _Float16 h_lds[64*256];
  for (int i = tid; i < 64*256; i += 1024) h_lds[i] = (_Float16)0.0f;

  float cstate[2][8];
  float bsum[2][4];
#pragma unroll
  for (int pp = 0; pp < 2; ++pp) {
    int pair = (wave << 1) + pp;
    int utile = pair & 15;
    int n_u = utile*16 + (lane & 15);
#pragma unroll
    for (int g = 0; g < 4; ++g) bsum[pp][g] = bi[g*256 + n_u] + bh[g*256 + n_u];
#pragma unroll
    for (int r = 0; r < 8; ++r) cstate[pp][r] = 0.0f;
  }
  __syncthreads();

  const size_t bstep = (size_t)64 * 512;   // one k-tile of packed B (ntiles=64)

  for (int tau = 0; tau < 25; ++tau) {
    const int tx = dir ? tau : 24 - tau;
    float hout[2][8];

#pragma unroll
    for (int pp = 0; pp < 2; ++pp) {
      int pair  = (wave << 1) + pp;
      int mtile = pair >> 4;
      int utile = pair & 15;

      v8f acc0 = {}, acc1 = {}, acc2 = {}, acc3 = {};

      // input contribution: K = 320 (300 + zero pad)
      {
        const _Float16* ap = r_in_h + ((size_t)(mtile*16 + (lane & 15))*25 + tx)*320
                                    + ((lane >> 4) << 3);
        const _Float16* bp0 = b_base(Bi, lane, 0*16 + utile, 64);
        const _Float16* bp1 = b_base(Bi, lane, 1*16 + utile, 64);
        const _Float16* bp2 = b_base(Bi, lane, 2*16 + utile, 64);
        const _Float16* bp3 = b_base(Bi, lane, 3*16 + utile, 64);
        for (int kt = 0; kt < 10; ++kt) {
          v16h a = frag_a16(ap);
          acc0 = WMMA(a, frag_b(bp0), acc0);
          acc1 = WMMA(a, frag_b(bp1), acc1);
          acc2 = WMMA(a, frag_b(bp2), acc2);
          acc3 = WMMA(a, frag_b(bp3), acc3);
          ap += 32; bp0 += bstep; bp1 += bstep; bp2 += bstep; bp3 += bstep;
        }
      }
      // recurrent contribution: K = 256
      {
        const _Float16* ap = h_lds + (mtile*16 + (lane & 15))*256 + ((lane >> 4) << 3);
        const _Float16* bp0 = b_base(Bh, lane, 0*16 + utile, 64);
        const _Float16* bp1 = b_base(Bh, lane, 1*16 + utile, 64);
        const _Float16* bp2 = b_base(Bh, lane, 2*16 + utile, 64);
        const _Float16* bp3 = b_base(Bh, lane, 3*16 + utile, 64);
        for (int kt = 0; kt < 8; ++kt) {
          v16h a = frag_a16(ap);
          acc0 = WMMA(a, frag_b(bp0), acc0);
          acc1 = WMMA(a, frag_b(bp1), acc1);
          acc2 = WMMA(a, frag_b(bp2), acc2);
          acc3 = WMMA(a, frag_b(bp3), acc3);
          ap += 32; bp0 += bstep; bp1 += bstep; bp2 += bstep; bp3 += bstep;
        }
      }
#pragma unroll
      for (int r = 0; r < 8; ++r) {
        float gi = acc0[r] + bsum[pp][0];
        float gf = acc1[r] + bsum[pp][1];
        float gg = acc2[r] + bsum[pp][2];
        float go = acc3[r] + bsum[pp][3];
        float c  = sigf(gf) * cstate[pp][r] + sigf(gi) * tanhf(gg);
        cstate[pp][r] = c;
        hout[pp][r] = sigf(go) * tanhf(c);
      }
    }

    __syncthreads();   // all waves done reading h_lds for this step
#pragma unroll
    for (int pp = 0; pp < 2; ++pp) {
      int pair  = (wave << 1) + pp;
      int mtile = pair >> 4;
      int utile = pair & 15;
      int u = utile*16 + (lane & 15);
#pragma unroll
      for (int r = 0; r < 8; ++r) {
        int m = mtile*16 + r + ((lane >> 4) << 3);
        _Float16 hv = (_Float16)hout[pp][r];
        h_lds[m*256 + u] = hv;
        hs[((size_t)m*25 + tx)*512 + (dir << 8) + u] = hv;
      }
    }
    __syncthreads();   // new h visible before next step's reads
  }
}

// ---------------------------------------------------------------------------
// Output projection: sigmoid(hs[1600,512] @ h2o_w^T[512,31] + b).
// 100 m-tiles x 2 n-tiles; one wave per n-tile (2 waves/block).
// Dual accumulator chains over the 16 k-tiles.
// ---------------------------------------------------------------------------
__global__ __launch_bounds__(64) void out_kernel(
    const _Float16* __restrict__ hs, const _Float16* __restrict__ Bo,
    const float* __restrict__ bo, float* __restrict__ out)
{
  const int wave = threadIdx.x >> 5, lane = threadIdx.x & 31;
  const int mtile = blockIdx.x;

  const _Float16* ap = hs + (size_t)(mtile*16 + (lane & 15))*512 + ((lane >> 4) << 3);
  const _Float16* bp = b_base(Bo, lane, wave, 2);
  const size_t bstep = (size_t)2 * 512;

  v8f accA = {}, accB = {};
  for (int kt = 0; kt < 16; kt += 2) {
    v16h a0 = frag_a16(ap);
    v16h b0 = frag_b(bp);
    v16h a1 = frag_a16(ap + 32);
    v16h b1 = frag_b(bp + bstep);
    accA = WMMA(a0, b0, accA);
    accB = WMMA(a1, b1, accB);
    ap += 64; bp += 2*bstep;
  }

  int n = (wave << 4) + (lane & 15);
  if (n < 31) {
    float bb = bo[n];
#pragma unroll
    for (int r = 0; r < 8; ++r) {
      int m = mtile*16 + r + ((lane >> 4) << 3);
      out[(size_t)m*31 + n] = sigf(accA[r] + accB[r] + bb);
    }
  }
}

// ---------------------------------------------------------------------------
extern "C" void kernel_launch(void* const* d_in, const int* in_sizes, int n_in,
                              void* d_out, int out_size, void* d_ws, size_t ws_size,
                              hipStream_t stream) {
  (void)in_sizes; (void)n_in; (void)out_size; (void)ws_size;
  const int*   dialogue = (const int*)  d_in[0];
  const float* emb      = (const float*)d_in[1];
  const float* w3 = (const float*)d_in[2];
  const float* b3 = (const float*)d_in[3];
  const float* w4 = (const float*)d_in[4];
  const float* b4 = (const float*)d_in[5];
  const float* w5 = (const float*)d_in[6];
  const float* b5 = (const float*)d_in[7];
  const float* wih  = (const float*)d_in[8];
  const float* whh  = (const float*)d_in[9];
  const float* bih  = (const float*)d_in[10];
  const float* bhh  = (const float*)d_in[11];
  const float* wihr = (const float*)d_in[12];
  const float* whhr = (const float*)d_in[13];
  const float* bihr = (const float*)d_in[14];
  const float* bhhr = (const float*)d_in[15];
  const float* h2ow = (const float*)d_in[16];
  const float* h2ob = (const float*)d_in[17];
  float* out = (float*)d_out;

  // workspace carve-up (f16 elements; every array size is a multiple of 512)
  _Float16* p = (_Float16*)d_ws;
  _Float16* B3   = p; p += 29*7*512;     // fs=3: K=900 -> 29 ktiles, 7 ntiles
  _Float16* B4   = p; p += 38*7*512;     // fs=4: K=1200 -> 38 ktiles
  _Float16* B5   = p; p += 47*7*512;     // fs=5: K=1500 -> 47 ktiles
  _Float16* Bih0 = p; p += 10*64*512;    // K=300->320, N=1024
  _Float16* Bhh0 = p; p += 8*64*512;     // K=256, N=1024
  _Float16* Bih1 = p; p += 10*64*512;
  _Float16* Bhh1 = p; p += 8*64*512;
  _Float16* Bo   = p; p += 16*2*512;     // K=512, N=31->32
  _Float16* rinh = p; p += (size_t)1600*320;
  _Float16* hsh  = p; p += (size_t)1600*512;

  auto prep = [&](const float* s, _Float16* d, int N, int K, int kt, int nt){
    int total = kt*nt*512;
    prep_b_packed<<<(total + 255)/256, 256, 0, stream>>>(s, d, N, K, kt, nt);
  };
  prep(w3,   B3,   100,  900, 29, 7);
  prep(w4,   B4,   100, 1200, 38, 7);
  prep(w5,   B5,   100, 1500, 47, 7);
  prep(wih,  Bih0, 1024, 300, 10, 64);
  prep(whh,  Bhh0, 1024, 256,  8, 64);
  prep(wihr, Bih1, 1024, 300, 10, 64);
  prep(whhr, Bhh1, 1024, 256,  8, 64);
  prep(h2ow, Bo,     31, 512, 16, 2);

  conv_embed_kernel<<<1600, 256, 0, stream>>>(dialogue, emb, B3, B4, B5, b3, b4, b5, rinh);
  lstm_kernel<<<2, 1024, 0, stream>>>(rinh, Bih0, Bhh0, bih, bhh, Bih1, Bhh1, bihr, bhhr, hsh);
  out_kernel<<<100, 64, 0, stream>>>(hsh, Bo, h2ob, out);
}
